// FViTBackbone_21998822490777
// MI455X (gfx1250) — matrix operand order repaired and checked by reading, not compiled, not run on hardware
//
#include <hip/hip_runtime.h>
#include <hip/hip_bf16.h>

typedef __attribute__((ext_vector_type(16))) _Float16 v16h;
typedef __attribute__((ext_vector_type(8)))  float    v8f;
// 16-byte payload type for the async-to-LDS builtin (<4 x i32>), per hipcc's
// diagnostic: param is `__vector_size__(4*sizeof(int)) int *` in AS(1)/AS(3).
typedef __attribute__((__vector_size__(4 * sizeof(int)))) int b128_t;

#if __has_builtin(__builtin_amdgcn_global_load_async_to_lds_b128) && \
    __has_builtin(__builtin_amdgcn_s_wait_asynccnt)
#define USE_ASYNC_LDS 1
#else
#define USE_ASYNC_LDS 0
#endif

// LDS pitch (in halfs) for the conv B tile ([token][K] layout, 96-byte rows:
// keeps every 16-half fragment 32-byte aligned).
#define BPITCH 48

// ---------------------------------------------------------------------------
// WMMA GEMM: Y[n][o][t] (+=) sum_k W[o][k] * Xh[n][k][t] + bias[o]
// Xh is f16; W/bias f32; Y is OutT (f16 feed-forward tensors, f32 residual).
// Block = 256 threads (8 waves) computes a 128(M) x 64(N) tile.
// The f16 B tile (32K x 64N = 4 KB) is staged into LDS with
// GLOBAL_LOAD_ASYNC_TO_LDS_B128 (16 bytes/thread, ASYNCcnt-tracked), then
// shared by all 8 waves; each wave reuses its A fragment across 4 WMMAs.
// grid = (T/64, O/128, N)
// ---------------------------------------------------------------------------
template<bool RELU, bool ACCUM, typename OutT>
__global__ void gemm_wmma(const float* __restrict__ Wm, const _Float16* __restrict__ Xh,
                          const float* __restrict__ bias, OutT* __restrict__ Y,
                          int K, int T, int O) {
  __shared__ _Float16 bt[32 * 64];          // [k][n], rows of 128 B

  const int tid  = threadIdx.x;
  const int lane = tid & 31;
  const int wave = tid >> 5;
  const int m0  = blockIdx.y * 128 + wave * 16;
  const int gn0 = blockIdx.x * 64;
  const int hi  = lane >> 4;                // 0: lanes 0-15, 1: lanes 16-31
  const int row = m0 + (lane & 15);
  const _Float16* Xn = Xh + (size_t)blockIdx.z * (size_t)K * (size_t)T;
  OutT* Yn = Y + (size_t)blockIdx.z * (size_t)O * (size_t)T;

  // staging assignment: thread copies 8 halfs (16 B) of row k
  const int sk = tid >> 3;                  // 0..31
  const int sn = (tid & 7) << 3;            // 0,8,..,56

  v8f acc[4] = {};
  for (int k0 = 0; k0 < K; k0 += 32) {
    // ---- stage B tile: global f16 -> LDS, no conversion needed ----
    {
      const _Float16* gsrc = Xn + (size_t)(k0 + sk) * T + gn0 + sn;
      _Float16* ldst = &bt[sk * 64 + sn];
#if USE_ASYNC_LDS
      __builtin_amdgcn_global_load_async_to_lds_b128(
          (__attribute__((address_space(1))) b128_t*)gsrc,
          (__attribute__((address_space(3))) b128_t*)ldst, 0, 0);
#else
      *reinterpret_cast<uint4*>(ldst) = *reinterpret_cast<const uint4*>(gsrc);
#endif
    }
#if USE_ASYNC_LDS
    __builtin_amdgcn_s_wait_asynccnt(0);
#endif
    __syncthreads();

    // ---- A fragment: row-major f32 weight, one contiguous 16-float run ----
    v16h a;
    const float* wrow = Wm + (size_t)row * K + k0 + hi * 16;
#pragma unroll
    for (int j = 0; j < 8; ++j) {
      a[j]     = (_Float16)wrow[j];
      a[j + 8] = (_Float16)wrow[8 + j];
    }

    // ---- 4 WMMAs sharing the A fragment ----
    const int kb = hi * 16;
#pragma unroll
    for (int nt = 0; nt < 4; ++nt) {
      const int coll = nt * 16 + (lane & 15);
      v16h b;
#pragma unroll
      for (int j = 0; j < 16; ++j)
        b[j] = bt[(kb + j) * 64 + coll];    // lanes hit consecutive 2B addrs
      acc[nt] = __builtin_amdgcn_wmma_f32_16x16x32_f16(false, a, false, b,
                                                       (short)0, acc[nt], false, false);
    }
    __syncthreads();
  }

#pragma unroll
  for (int nt = 0; nt < 4; ++nt) {
    const int col = gn0 + nt * 16 + (lane & 15);
#pragma unroll
    for (int i = 0; i < 8; ++i) {
      const int m = m0 + i + hi * 8;
      float v = acc[nt][i] + bias[m];
      if (ACCUM) v += (float)Yn[(size_t)m * T + col];
      if (RELU)  v = fmaxf(v, 0.f);
      Yn[(size_t)m * T + col] = (OutT)v;
    }
  }
}

// ---------------------------------------------------------------------------
// Implicit-GEMM 3x3 stride-2 pad-1 conv via WMMA.  K = Cin*9.  f32 in/out.
// The im2col B tile (32K x 16 tokens) is gathered cooperatively ONCE per
// block into LDS (decode + OOB check once), then shared by all 8 waves.
// FUSED_EMBED: input is the 3-channel image; the 1x1 embed (ew,eb) is applied
// inside the gather (zero padding applies to the embed output).
// grid = (OH*OW/16, O/128, N)
// ---------------------------------------------------------------------------
template<bool FUSED_EMBED>
__global__ void conv3x3s2_wmma(const float* __restrict__ Wm, const float* __restrict__ X,
                               const float* __restrict__ bias, float* __restrict__ Y,
                               int Cin, int H, int Wd, int OH, int OW, int O,
                               const float* __restrict__ ew, const float* __restrict__ eb) {
  __shared__ _Float16 bh[16 * BPITCH];

  const int tid  = threadIdx.x;
  const int lane = tid & 31;
  const int wave = tid >> 5;
  const int m0 = blockIdx.y * 128 + wave * 16;
  const int n0 = blockIdx.x * 16;
  const int hi = lane >> 4;
  const int row = m0 + (lane & 15);
  const int K = Cin * 9;
  const float* Xn = X + (size_t)blockIdx.z * (size_t)(FUSED_EMBED ? 3 : Cin) * H * Wd;
  const size_t hw = (size_t)H * (size_t)Wd;

  v8f acc = {};
  for (int k0 = 0; k0 < K; k0 += 32) {
#pragma unroll
    for (int e = 0; e < 2; ++e) {
      const int idx = tid + e * 256;        // 0..511
      const int tt = idx & 15;
      const int k  = idx >> 4;              // 0..31
      const int t  = n0 + tt;
      const int oh = t / OW, ow = t - (t / OW) * OW;
      const int kk = k0 + k;
      const int c  = kk / 9;
      const int rs = kk - c * 9;
      const int r  = rs / 3;
      const int s  = rs - r * 3;
      const int ih = oh * 2 - 1 + r;
      const int iw = ow * 2 - 1 + s;
      float v = 0.f;
      if ((unsigned)ih < (unsigned)H && (unsigned)iw < (unsigned)Wd) {
        if (FUSED_EMBED) {
          const size_t base = (size_t)ih * Wd + iw;
          v = eb[c] + ew[c * 3 + 0] * Xn[base]
                    + ew[c * 3 + 1] * Xn[hw + base]
                    + ew[c * 3 + 2] * Xn[2 * hw + base];
        } else {
          v = Xn[((size_t)c * H + ih) * Wd + iw];
        }
      }
      bh[tt * BPITCH + k] = (_Float16)v;
    }
    __syncthreads();

    v16h a;
    const float* wrow = Wm + (size_t)row * K + k0 + hi * 16;
#pragma unroll
    for (int j = 0; j < 8; ++j) {
      a[j]     = (_Float16)wrow[j];
      a[j + 8] = (_Float16)wrow[8 + j];
    }
    const v16h b = *reinterpret_cast<const v16h*>(&bh[(lane & 15) * BPITCH + hi * 16]);
    acc = __builtin_amdgcn_wmma_f32_16x16x32_f16(false, a, false, b,
                                                 (short)0, acc, false, false);
    __syncthreads();
  }

  const int t = n0 + (lane & 15);
  float* Yn = Y + (size_t)blockIdx.z * (size_t)O * (size_t)OH * (size_t)OW;
#pragma unroll
  for (int i = 0; i < 8; ++i) {
    const int m = m0 + i + hi * 8;
    Yn[(size_t)m * (OH * OW) + t] = acc[i] + bias[m];
  }
}

// ---------------------------------------------------------------------------
// InstanceNorm (affine=False) per (n,c) plane; optional ReLU; OutT output.
// grid = N*C.
// ---------------------------------------------------------------------------
template<bool RELU, typename OutT>
__global__ void inorm_kernel(const float* __restrict__ X, OutT* __restrict__ Y, int HW) {
  const float* p = X + (size_t)blockIdx.x * HW;
  OutT* q = Y + (size_t)blockIdx.x * HW;
  float s = 0.f, ss = 0.f;
  for (int i = threadIdx.x; i < HW; i += blockDim.x) {
    const float v = p[i];
    s += v; ss += v * v;
  }
  for (int off = 16; off; off >>= 1) {
    s  += __shfl_xor(s,  off, 32);
    ss += __shfl_xor(ss, off, 32);
  }
  __shared__ float red[2][8];
  const int lane = threadIdx.x & 31, wave = threadIdx.x >> 5;
  if (lane == 0) { red[0][wave] = s; red[1][wave] = ss; }
  __syncthreads();
  if (threadIdx.x == 0) {
    float a = 0.f, b = 0.f;
    for (int i = 0; i < 8; ++i) { a += red[0][i]; b += red[1][i]; }
    red[0][0] = a; red[1][0] = b;
  }
  __syncthreads();
  const float mean = red[0][0] / (float)HW;
  const float var  = red[1][0] / (float)HW - mean * mean;
  const float inv  = rsqrtf(var + 1e-5f);
  for (int i = threadIdx.x; i < HW; i += blockDim.x) {
    float v = (p[i] - mean) * inv;
    if (RELU) v = fmaxf(v, 0.f);
    q[i] = (OutT)v;
  }
}

// ---------------------------------------------------------------------------
// Axial attention core (LDS resident, f16 in/out).  One block per
// (batch-line, head).  alongW=1: sequence along W; alongW=0: along H.
// qkv is NCHW with 3C channels; ctx written NCHW with C channels.
// ---------------------------------------------------------------------------
__global__ void attn_kernel(const _Float16* __restrict__ qkv, _Float16* __restrict__ ctx,
                            int N, int C, int H, int Wd, int heads, int d, int alongW) {
  const int S = alongW ? Wd : H;
  const int b = blockIdx.x;
  const int head = blockIdx.y;
  const int n   = alongW ? (b / H)  : (b / Wd);
  const int fix = alongW ? (b % H)  : (b % Wd);

  __shared__ float q[64][33], k[64][33], v[64][33];
  __shared__ float sc[64][65];

  const int tid = threadIdx.x;
  const int C3 = 3 * C;
  for (int i = tid; i < S * d; i += blockDim.x) {
    const int s = i / d, c = i - (i / d) * d;
    const int h = alongW ? fix : s;
    const int w = alongW ? s : fix;
    const size_t base = (((size_t)n * C3) * H + h) * Wd + w;
    const size_t chs  = (size_t)H * Wd;   // channel stride
    q[s][c] = (float)qkv[base + (size_t)(head * d + c) * chs];
    k[s][c] = (float)qkv[base + (size_t)(C + head * d + c) * chs];
    v[s][c] = (float)qkv[base + (size_t)(2 * C + head * d + c) * chs];
  }
  __syncthreads();

  const float scale = rsqrtf((float)d);
  for (int i = tid; i < S * S; i += blockDim.x) {
    const int r = i / S, cc = i - (i / S) * S;
    float acc = 0.f;
    for (int e = 0; e < d; ++e) acc += q[r][e] * k[cc][e];
    sc[r][cc] = acc * scale;
  }
  __syncthreads();

  for (int r = tid; r < S; r += blockDim.x) {
    float mx = -1e30f;
    for (int j = 0; j < S; ++j) mx = fmaxf(mx, sc[r][j]);
    float sum = 0.f;
    for (int j = 0; j < S; ++j) { const float e = __expf(sc[r][j] - mx); sc[r][j] = e; sum += e; }
    const float inv = 1.f / sum;
    for (int j = 0; j < S; ++j) sc[r][j] *= inv;
  }
  __syncthreads();

  for (int i = tid; i < S * d; i += blockDim.x) {
    const int s = i / d, c = i - (i / d) * d;
    float acc = 0.f;
    for (int j = 0; j < S; ++j) acc += sc[s][j] * v[j][c];
    const int h = alongW ? fix : s;
    const int w = alongW ? s : fix;
    ctx[(((size_t)n * C + head * d + c) * H + h) * Wd + w] = (_Float16)acc;
  }
}

// x[n][c][hw] += pos[c][hw]  (pos broadcast over batch)
__global__ void add_pos_kernel(float* __restrict__ x, const float* __restrict__ pos,
                               int total, int chw) {
  const int i = blockIdx.x * blockDim.x + threadIdx.x;
  if (i < total) x[i] += pos[i % chw];
}

// ---------------------------------------------------------------------------
// Host side
// ---------------------------------------------------------------------------
struct BlockP {
  const float *h_inb, *h_inw, *h_outb, *h_outw;
  const float *mb1, *mb2, *mw1, *mw2;
  const float *v_inb, *v_inw, *v_outb, *v_outw;
};

extern "C" void kernel_launch(void* const* d_in, const int* in_sizes, int n_in,
                              void* d_out, int out_size, void* d_ws, size_t ws_size,
                              hipStream_t stream) {
  (void)in_sizes; (void)n_in; (void)out_size; (void)ws_size;
  const int N = 8;

  // JAX pytree (alphabetical dict-key) flattening of setup_inputs():
  // images, embed{downs[(w,b)x3], in_b, in_w, pos}, stages[{blocks[...], proj_b, proj_w} x2]
  const float* img  = (const float*)d_in[0];
  const float* dw0  = (const float*)d_in[1];
  const float* db0  = (const float*)d_in[2];
  const float* dw1  = (const float*)d_in[3];
  const float* db1  = (const float*)d_in[4];
  const float* dw2  = (const float*)d_in[5];
  const float* db2  = (const float*)d_in[6];
  const float* e_ib = (const float*)d_in[7];   // in_b (128)
  const float* e_iw = (const float*)d_in[8];   // in_w (128,3,1,1)
  const float* pos  = (const float*)d_in[9];   // (1,128,64,64)

  BlockP blk[2][2];
  const float* proj_b[2];
  const float* proj_w[2];
  int ix = 10;
  for (int s = 0; s < 2; ++s) {
    for (int bb = 0; bb < 2; ++bb) {
      BlockP& p = blk[s][bb];
      p.h_inb  = (const float*)d_in[ix++];
      p.h_inw  = (const float*)d_in[ix++];
      p.h_outb = (const float*)d_in[ix++];
      p.h_outw = (const float*)d_in[ix++];
      p.mb1    = (const float*)d_in[ix++];
      p.mb2    = (const float*)d_in[ix++];
      p.mw1    = (const float*)d_in[ix++];
      p.mw2    = (const float*)d_in[ix++];
      p.v_inb  = (const float*)d_in[ix++];
      p.v_inw  = (const float*)d_in[ix++];
      p.v_outb = (const float*)d_in[ix++];
      p.v_outw = (const float*)d_in[ix++];
    }
    proj_b[s] = (const float*)d_in[ix++];
    proj_w[s] = (const float*)d_in[ix++];
  }

  // Workspace carve (byte cursor, 256-B aligned blocks)
  char* wsb = (char*)d_ws;
  size_t cur = 0;
  auto carve = [&](size_t bytes) {
    void* p = wsb + cur;
    cur += (bytes + 255) & ~(size_t)255;
    return p;
  };
  float*    X0   = (float*)   carve((size_t)N * 128 * 256 * 256 * 4);  // f32 ping
  float*    X1   = (float*)   carve((size_t)N * 128 * 256 * 256 * 4);  // f32 pong
  _Float16* QKVV = (_Float16*)carve((size_t)N * 384 * 64 * 64 * 2);
  _Float16* QKVH = (_Float16*)carve((size_t)N * 384 * 64 * 64 * 2);
  _Float16* CTXV = (_Float16*)carve((size_t)N * 128 * 64 * 64 * 2);
  _Float16* CTXH = (_Float16*)carve((size_t)N * 128 * 64 * 64 * 2);
  _Float16* XN   = (_Float16*)carve((size_t)N * 128 * 64 * 64 * 2);
  _Float16* T1   = (_Float16*)carve((size_t)N * 512 * 64 * 64 * 2);

  const dim3 B(256);

  // ---- Embed + downsample chain (embed 1x1 fused into first conv) ----
  conv3x3s2_wmma<true ><<<dim3(256 * 256 / 16, 1, N), B, 0, stream>>>(
      dw0, img, db0, X1, 128, 512, 512, 256, 256, 128, e_iw, e_ib);
  inorm_kernel<true, float><<<N * 128, 256, 0, stream>>>(X1, X0, 256 * 256);

  conv3x3s2_wmma<false><<<dim3(128 * 128 / 16, 1, N), B, 0, stream>>>(
      dw1, X0, db1, X1, 128, 256, 256, 128, 128, 128, nullptr, nullptr);
  inorm_kernel<true, float><<<N * 128, 256, 0, stream>>>(X1, X0, 128 * 128);

  conv3x3s2_wmma<false><<<dim3(64 * 64 / 16, 1, N), B, 0, stream>>>(
      dw2, X0, db2, X1, 128, 128, 128, 64, 64, 128, nullptr, nullptr);
  inorm_kernel<true, float><<<N * 128, 256, 0, stream>>>(X1, X0, 64 * 64);

  add_pos_kernel<<<(N * 128 * 64 * 64 + 255) / 256, 256, 0, stream>>>(
      X0, pos, N * 128 * 64 * 64, 128 * 64 * 64);

  // ---- Transformer block runner ----
  auto run_block = [&](float* x, const BlockP& p, int C, int Hh, int Ww, int heads) {
    const int HW = Hh * Ww;         // 4096 (stage0) / 1024 (stage1); both %64==0
    const int d = C / heads;
    // attention branch
    inorm_kernel<false, _Float16><<<N * C, 256, 0, stream>>>(x, XN, HW);
    gemm_wmma<false, false, _Float16><<<dim3(HW / 64, 3 * C / 128, N), B, 0, stream>>>(
        p.v_inw, XN, p.v_inb, QKVV, C, HW, 3 * C);
    gemm_wmma<false, false, _Float16><<<dim3(HW / 64, 3 * C / 128, N), B, 0, stream>>>(
        p.h_inw, XN, p.h_inb, QKVH, C, HW, 3 * C);
    attn_kernel<<<dim3(N * Hh, heads), 256, 0, stream>>>(QKVV, CTXV, N, C, Hh, Ww, heads, d, 1);
    attn_kernel<<<dim3(N * Ww, heads), 256, 0, stream>>>(QKVH, CTXH, N, C, Hh, Ww, heads, d, 0);
    gemm_wmma<false, true, float><<<dim3(HW / 64, C / 128, N), B, 0, stream>>>(
        p.v_outw, CTXV, p.v_outb, x, C, HW, C);   // x += out_v
    gemm_wmma<false, true, float><<<dim3(HW / 64, C / 128, N), B, 0, stream>>>(
        p.h_outw, CTXH, p.h_outb, x, C, HW, C);   // x += out_h
    // MLP branch
    inorm_kernel<false, _Float16><<<N * C, 256, 0, stream>>>(x, XN, HW);
    gemm_wmma<true, false, _Float16><<<dim3(HW / 64, 4 * C / 128, N), B, 0, stream>>>(
        p.mw1, XN, p.mb1, T1, C, HW, 4 * C);
    gemm_wmma<false, true, float><<<dim3(HW / 64, C / 128, N), B, 0, stream>>>(
        p.mw2, T1, p.mb2, x, 4 * C, HW, C);       // x += mlp
  };

  // ---- Stage 0: C=128, 64x64 ----
  run_block(X0, blk[0][0], 128, 64, 64, 8);
  run_block(X0, blk[0][1], 128, 64, 64, 8);
  conv3x3s2_wmma<false><<<dim3(32 * 32 / 16, 256 / 128, N), B, 0, stream>>>(
      proj_w[0], X0, proj_b[0], X1, 128, 64, 64, 32, 32, 256, nullptr, nullptr);

  // ---- Stage 1: C=256, 32x32 ----
  run_block(X1, blk[1][0], 256, 32, 32, 8);
  run_block(X1, blk[1][1], 256, 32, 32, 8);
  conv3x3s2_wmma<false><<<dim3(16 * 16 / 16, 512 / 128, N), B, 0, stream>>>(
      proj_w[1], X1, proj_b[1], (float*)d_out, 256, 32, 32, 16, 16, 512, nullptr, nullptr);
}
